// MultiTaskGNN_13477607375565
// MI455X (gfx1250) — matrix operand order repaired and checked by reading, not compiled, not run on hardware
//
#include <hip/hip_runtime.h>
#include <hip/hip_bf16.h>
#include <math.h>

// ---------------------------------------------------------------------------
// Problem constants (match reference)
// ---------------------------------------------------------------------------
#define NN    50000
#define EE    400000
#define ET    (EE + NN)      // edges + self loops
#define DIN   256
#define HID   64
#define HEADS 4
#define F1    (HEADS * HID)  // 256
#define KT8   8              // K = 256 -> 8 steps of 32
#define NTG   4              // column tiles per wave (register blocking)

typedef __attribute__((ext_vector_type(16))) __bf16 v16bf;
typedef __attribute__((ext_vector_type(8)))  float  v8f;

// ---------------------------------------------------------------------------
// Workspace layout (float offsets)
// ---------------------------------------------------------------------------
#define O_H1    ((size_t)0)                     // N*256  raw h1 = x@W1
#define O_AGG1  (O_H1   + (size_t)NN*F1)        // N*256  aggregated (then act1 in-place)
#define O_AS1   (O_AGG1 + (size_t)NN*F1)        // N*4
#define O_AD1   (O_AS1  + (size_t)NN*HEADS)     // N*4
#define O_M1    (O_AD1  + (size_t)NN*HEADS)     // N*4
#define O_D1    (O_M1   + (size_t)NN*HEADS)     // N*4
#define O_H2    (O_D1   + (size_t)NN*HEADS)     // N*64  raw h2 = act1@W2
#define O_AGG2  (O_H2   + (size_t)NN*HID)       // N*64  aggregated (then act2 in-place)
#define O_AS2   (O_AGG2 + (size_t)NN*HID)       // N
#define O_AD2   (O_AS2  + (size_t)NN)           // N
#define O_M2    (O_AD2  + (size_t)NN)           // N
#define O_D2    (O_M2   + (size_t)NN)           // N
#define O_PW1   (O_D2   + (size_t)NN)           // packed W1: 8*16*512 bf16 = 32768 floats
#define O_PW2   (O_PW1  + (size_t)32768)        // packed W2: 8*4*512 bf16  = 8192 floats

// ---------------------------------------------------------------------------
// Pack W (row-major [K, Ncols] f32) into WMMA B-fragment order, bf16.
// Fragment (kt, nt): 32 lanes x 16 bf16 contiguous per lane.
//   lane < 16 : col = nt*16+lane,     K = kt*32 + (j<8 ? j : j+8)       (j = 0..15)
//   lane >= 16: col = nt*16+lane-16,  K = kt*32 + 8 + (j<8 ? j : j+8)
// ---------------------------------------------------------------------------
__global__ void pack_w_bf16(const float* __restrict__ W, __bf16* __restrict__ pW,
                            int KT, int NT, int ldw) {
    int tid   = blockIdx.x * blockDim.x + threadIdx.x;
    int total = KT * NT * 512;
    if (tid >= total) return;
    int j    = tid & 15;
    int lane = (tid >> 4) & 31;
    int frag = tid >> 9;
    int nt   = frag % NT;
    int kt   = frag / NT;
    int k    = kt * 32 + ((lane >= 16) ? 8 : 0) + ((j < 8) ? j : j + 8);
    int n    = nt * 16 + (lane & 15);
    pW[tid]  = (__bf16)W[(size_t)k * ldw + n];
}

// ---------------------------------------------------------------------------
// GEMM: C[M,Ncols] = A[M,K] * B[K,Ncols]; A fp32 row-major, B pre-packed bf16.
// Block = 8 waves. Each block stages one 4-column group of B fragments (all
// of K) into LDS (32 KB), then each wave computes a 16x64 output stripe:
// per kt, one A-fragment build feeds 4 v_wmma_f32_16x16x32_bf16.
// Grid = ceil(MT/8) * (NT/NTG).
// ---------------------------------------------------------------------------
__global__ void __launch_bounds__(256)
gemm_wmma_bf16(const float* __restrict__ A, const __bf16* __restrict__ pB,
               float* __restrict__ C, int MT, int NT, int lda, int ldc) {
    __shared__ __bf16 lB[KT8 * NTG * 512];     // 32 KB

    const int nGroups = NT / NTG;
    const int rg      = blockIdx.x / nGroups;
    const int ng      = blockIdx.x % nGroups;
    const int waveId  = threadIdx.x >> 5;
    const int lane    = threadIdx.x & 31;

    // cooperative stage of KT8*NTG fragments (1024 B each, contiguous)
    {
        uint4*      dst = (uint4*)lB;
        const int   totalVec = KT8 * NTG * 64;                 // 16B vectors
        for (int i = threadIdx.x; i < totalVec; i += blockDim.x) {
            int f   = i >> 6;                                  // fragment id
            int off = i & 63;
            int kt  = f / NTG, j = f - kt * NTG;
            const uint4* s =
                (const uint4*)(pB + ((size_t)(kt * NT + ng * NTG + j)) * 512);
            dst[i] = s[off];
        }
    }
    __syncthreads();

    const int mt = rg * 8 + waveId;
    if (mt < MT) {                                 // wave-uniform: EXEC stays full
        const int off = (lane < 16) ? 0 : 8;       // A-fragment K offset (ISA layout)
        const int row = mt * 16 + (lane & 15);
        const float* rp = A + (size_t)row * lda + off;

        v8f c0 = {0.f,0.f,0.f,0.f,0.f,0.f,0.f,0.f};
        v8f c1 = c0, c2 = c0, c3 = c0;

        for (int kt = 0; kt < KT8; ++kt) {
            const float* p = rp + kt * 32;
            __builtin_prefetch(p + 32, 0, 3);      // global_prefetch_b8 next K-slab
            float4 f0 = *(const float4*)(p);
            float4 f1 = *(const float4*)(p + 4);
            float4 f2 = *(const float4*)(p + 16);
            float4 f3 = *(const float4*)(p + 20);
            v16bf a;
            a[0]=(__bf16)f0.x;  a[1]=(__bf16)f0.y;  a[2]=(__bf16)f0.z;  a[3]=(__bf16)f0.w;
            a[4]=(__bf16)f1.x;  a[5]=(__bf16)f1.y;  a[6]=(__bf16)f1.z;  a[7]=(__bf16)f1.w;
            a[8]=(__bf16)f2.x;  a[9]=(__bf16)f2.y;  a[10]=(__bf16)f2.z; a[11]=(__bf16)f2.w;
            a[12]=(__bf16)f3.x; a[13]=(__bf16)f3.y; a[14]=(__bf16)f3.z; a[15]=(__bf16)f3.w;

            const __bf16* lbp = lB + (size_t)(kt * NTG) * 512 + (size_t)lane * 16;
            v16bf b0 = *(const v16bf*)(lbp);
            v16bf b1 = *(const v16bf*)(lbp + 512);
            v16bf b2 = *(const v16bf*)(lbp + 1024);
            v16bf b3 = *(const v16bf*)(lbp + 1536);
            c0 = __builtin_amdgcn_wmma_f32_16x16x32_bf16(false, a, false, b0,
                                                         (short)0, c0, false, false);
            c1 = __builtin_amdgcn_wmma_f32_16x16x32_bf16(false, a, false, b1,
                                                         (short)0, c1, false, false);
            c2 = __builtin_amdgcn_wmma_f32_16x16x32_bf16(false, a, false, b2,
                                                         (short)0, c2, false, false);
            c3 = __builtin_amdgcn_wmma_f32_16x16x32_bf16(false, a, false, b3,
                                                         (short)0, c3, false, false);
        }

        // C/D layout: VGPR r -> row mt*16 + r (+8 for lanes 16-31)
        const int roff = (lane < 16) ? 0 : 8;
        float* cbase = C + (size_t)(mt * 16 + roff) * ldc
                         + (size_t)(ng * NTG) * 16 + (lane & 15);
        #pragma unroll
        for (int r = 0; r < 8; ++r) {
            float* cp = cbase + (size_t)r * ldc;
            cp[0]  = c0[r];
            cp[16] = c1[r];
            cp[32] = c2[r];
            cp[48] = c3[r];
        }
    }
}

// ---------------------------------------------------------------------------
// Attention logits per node/head: as[n,h] = h[n,h,:].a_src[h,:], same for dst.
// ---------------------------------------------------------------------------
__global__ void alpha_kernel(const float* __restrict__ Hf,
                             const float* __restrict__ a_src,
                             const float* __restrict__ a_dst,
                             float* __restrict__ as, float* __restrict__ ad,
                             int NH, int Hh) {
    int tid = blockIdx.x * blockDim.x + threadIdx.x;
    if (tid >= NH) return;
    int n = tid / Hh, h = tid % Hh;
    const float* hp = Hf + ((size_t)n * Hh + h) * HID;
    const float* sp = a_src + h * HID;
    const float* dp = a_dst + h * HID;
    float s = 0.f, d = 0.f;
    #pragma unroll 8
    for (int i = 0; i < HID; ++i) { float v = hp[i]; s += v * sp[i]; d += v * dp[i]; }
    as[tid] = s;
    ad[tid] = d;
}

__global__ void fill_kernel(float* __restrict__ p, float v, int n) {
    int tid = blockIdx.x * blockDim.x + threadIdx.x;
    if (tid < n) p[tid] = v;
}

__device__ __forceinline__ void edge_sd(const int* __restrict__ ei, int e,
                                        int& s, int& d) {
    if (e < EE) { s = ei[e]; d = ei[EE + e]; }
    else        { s = d = e - EE; }
}

__device__ __forceinline__ float leaky02(float v) { return v > 0.f ? v : 0.2f * v; }

// pass 1: segment max of leaky_relu(as[src]+ad[dst]) into m[dst,h]
__global__ void edge_max_kernel(const int* __restrict__ ei,
                                const float* __restrict__ as,
                                const float* __restrict__ ad,
                                float* __restrict__ m, int Hh) {
    int tid = blockIdx.x * blockDim.x + threadIdx.x;
    if (tid >= ET * Hh) return;
    int e = tid / Hh, h = tid % Hh;
    int s, d; edge_sd(ei, e, s, d);
    float v = leaky02(as[s * Hh + h] + ad[d * Hh + h]);
    atomicMax(&m[d * Hh + h], v);
}

// pass 2: denom[dst,h] += exp(e - m[dst,h])
__global__ void edge_sum_kernel(const int* __restrict__ ei,
                                const float* __restrict__ as,
                                const float* __restrict__ ad,
                                const float* __restrict__ m,
                                float* __restrict__ den, int Hh) {
    int tid = blockIdx.x * blockDim.x + threadIdx.x;
    if (tid >= ET * Hh) return;
    int e = tid / Hh, h = tid % Hh;
    int s, d; edge_sd(ei, e, s, d);
    float v = leaky02(as[s * Hh + h] + ad[d * Hh + h]);
    atomicAdd(&den[d * Hh + h], __expf(v - m[d * Hh + h]));
}

// pass 3: agg[dst,h,:] += alpha * Hf[src,h,:]
__global__ void edge_aggr_kernel(const int* __restrict__ ei,
                                 const float* __restrict__ as,
                                 const float* __restrict__ ad,
                                 const float* __restrict__ m,
                                 const float* __restrict__ den,
                                 const float* __restrict__ Hf,
                                 float* __restrict__ agg, int Hh) {
    int tid = blockIdx.x * blockDim.x + threadIdx.x;
    if (tid >= ET * Hh) return;
    int e = tid / Hh, h = tid % Hh;
    int s, d; edge_sd(ei, e, s, d);
    float v     = leaky02(as[s * Hh + h] + ad[d * Hh + h]);
    float alpha = __expf(v - m[d * Hh + h]) / (den[d * Hh + h] + 1e-16f);
    const float4* sp = (const float4*)(Hf + ((size_t)s * Hh + h) * HID);
    float*        dp = agg + ((size_t)d * Hh + h) * HID;
    #pragma unroll
    for (int i = 0; i < HID / 4; ++i) {
        float4 f = sp[i];
        atomicAdd(dp + 4 * i + 0, f.x * alpha);
        atomicAdd(dp + 4 * i + 1, f.y * alpha);
        atomicAdd(dp + 4 * i + 2, f.z * alpha);
        atomicAdd(dp + 4 * i + 3, f.w * alpha);
    }
}

// bias + batchnorm + elu, in place
__global__ void post_kernel(float* __restrict__ x, const float* __restrict__ bias,
                            const float* __restrict__ w, const float* __restrict__ b,
                            const float* __restrict__ mean, const float* __restrict__ var,
                            int total, int C) {
    int tid = blockIdx.x * blockDim.x + threadIdx.x;
    if (tid >= total) return;
    int c   = tid % C;
    float v = x[tid] + bias[c];
    v = (v - mean[c]) * rsqrtf(var[c] + 1e-5f) * w[c] + b[c];
    x[tid] = v > 0.f ? v : __expf(v) - 1.f;
}

// final heads: out[0:N) = h@Wr+br ; out[N:2N) = sigmoid(h@Wc+bc)
__global__ void head_kernel(const float* __restrict__ h,
                            const float* __restrict__ Wr, const float* __restrict__ br,
                            const float* __restrict__ Wc, const float* __restrict__ bc,
                            float* __restrict__ out, int n) {
    int tid = blockIdx.x * blockDim.x + threadIdx.x;
    if (tid >= n) return;
    const float* hp = h + (size_t)tid * HID;
    float r = 0.f, c = 0.f;
    #pragma unroll 8
    for (int i = 0; i < HID; ++i) { float v = hp[i]; r += v * Wr[i]; c += v * Wc[i]; }
    out[tid]     = r + br[0];
    out[n + tid] = 1.f / (1.f + __expf(-(c + bc[0])));
}

// ---------------------------------------------------------------------------
extern "C" void kernel_launch(void* const* d_in, const int* in_sizes, int n_in,
                              void* d_out, int out_size, void* d_ws, size_t ws_size,
                              hipStream_t stream) {
    const float* x     = (const float*)d_in[0];
    const int*   ei    = (const int*)  d_in[1];
    const float* W1    = (const float*)d_in[2];
    const float* asrc1 = (const float*)d_in[3];
    const float* adst1 = (const float*)d_in[4];
    const float* b1    = (const float*)d_in[5];
    const float* bn1w  = (const float*)d_in[6];
    const float* bn1b  = (const float*)d_in[7];
    const float* bn1m  = (const float*)d_in[8];
    const float* bn1v  = (const float*)d_in[9];
    const float* W2    = (const float*)d_in[10];
    const float* asrc2 = (const float*)d_in[11];
    const float* adst2 = (const float*)d_in[12];
    const float* b2    = (const float*)d_in[13];
    const float* bn2w  = (const float*)d_in[14];
    const float* bn2b  = (const float*)d_in[15];
    const float* bn2m  = (const float*)d_in[16];
    const float* bn2v  = (const float*)d_in[17];
    const float* Wr    = (const float*)d_in[18];
    const float* br    = (const float*)d_in[19];
    const float* Wc    = (const float*)d_in[20];
    const float* bc    = (const float*)d_in[21];
    float* out = (float*)d_out;
    float* ws  = (float*)d_ws;

    const int B = 256;
    auto blocks = [](long long n) { return (unsigned)((n + 255) / 256); };

    __bf16* pW1 = (__bf16*)(ws + O_PW1);
    __bf16* pW2 = (__bf16*)(ws + O_PW2);

    // Pack weights into WMMA B-fragment order (cheap, deterministic)
    pack_w_bf16<<<blocks(8 * 16 * 512), B, 0, stream>>>(W1, pW1, 8, 16, F1);
    pack_w_bf16<<<blocks(8 * 4 * 512),  B, 0, stream>>>(W2, pW2, 8, 4, HID);

    // -------- Layer 1 --------
    // H1 = x @ W1 : grid = ceil(3125/8) row-groups x 4 column-groups
    gemm_wmma_bf16<<<dim3(391 * 4), B, 0, stream>>>(
        x, pW1, ws + O_H1, 3125, 16, DIN, F1);

    alpha_kernel<<<blocks((long long)NN * HEADS), B, 0, stream>>>(
        ws + O_H1, asrc1, adst1, ws + O_AS1, ws + O_AD1, NN * HEADS, HEADS);

    fill_kernel<<<blocks(NN * HEADS), B, 0, stream>>>(ws + O_M1, -INFINITY, NN * HEADS);
    hipMemsetAsync(ws + O_D1,   0, (size_t)NN * HEADS * sizeof(float), stream);
    hipMemsetAsync(ws + O_AGG1, 0, (size_t)NN * F1    * sizeof(float), stream);

    edge_max_kernel<<<blocks((long long)ET * HEADS), B, 0, stream>>>(
        ei, ws + O_AS1, ws + O_AD1, ws + O_M1, HEADS);
    edge_sum_kernel<<<blocks((long long)ET * HEADS), B, 0, stream>>>(
        ei, ws + O_AS1, ws + O_AD1, ws + O_M1, ws + O_D1, HEADS);
    edge_aggr_kernel<<<blocks((long long)ET * HEADS), B, 0, stream>>>(
        ei, ws + O_AS1, ws + O_AD1, ws + O_M1, ws + O_D1,
        ws + O_H1, ws + O_AGG1, HEADS);

    post_kernel<<<blocks((long long)NN * F1), B, 0, stream>>>(
        ws + O_AGG1, b1, bn1w, bn1b, bn1m, bn1v, NN * F1, F1);

    // -------- Layer 2 --------
    // H2 = act1 @ W2 : grid = ceil(3125/8) row-groups x 1 column-group
    gemm_wmma_bf16<<<dim3(391), B, 0, stream>>>(
        ws + O_AGG1, pW2, ws + O_H2, 3125, 4, F1, HID);

    alpha_kernel<<<blocks(NN), B, 0, stream>>>(
        ws + O_H2, asrc2, adst2, ws + O_AS2, ws + O_AD2, NN, 1);

    fill_kernel<<<blocks(NN), B, 0, stream>>>(ws + O_M2, -INFINITY, NN);
    hipMemsetAsync(ws + O_D2,   0, (size_t)NN       * sizeof(float), stream);
    hipMemsetAsync(ws + O_AGG2, 0, (size_t)NN * HID * sizeof(float), stream);

    edge_max_kernel<<<blocks(ET), B, 0, stream>>>(
        ei, ws + O_AS2, ws + O_AD2, ws + O_M2, 1);
    edge_sum_kernel<<<blocks(ET), B, 0, stream>>>(
        ei, ws + O_AS2, ws + O_AD2, ws + O_M2, ws + O_D2, 1);
    edge_aggr_kernel<<<blocks(ET), B, 0, stream>>>(
        ei, ws + O_AS2, ws + O_AD2, ws + O_M2, ws + O_D2,
        ws + O_H2, ws + O_AGG2, 1);

    post_kernel<<<blocks((long long)NN * HID), B, 0, stream>>>(
        ws + O_AGG2, b2, bn2w, bn2b, bn2m, bn2v, NN * HID, HID);

    // -------- Heads --------
    head_kernel<<<blocks(NN), B, 0, stream>>>(ws + O_AGG2, Wr, br, Wc, bc, out, NN);
}
